// Snn_37958920962200
// MI455X (gfx1250) — compile-verified, matrix operand order checked by reading, not run on hardware
//
#include <hip/hip_runtime.h>

// ---------------------------------------------------------------------------
// CDNA5 / gfx1250 SNN forward.
//  - layer0 GEMM hoisted out of the time loop (x is static per step)
//  - layers 1/2: bf16 WMMA GEMM; spikes are exact in bf16, weights use
//    hi+lo bf16 split for ~fp32 accuracy; f32 accumulate in v_wmma.
//  - LIF fused into GEMM epilogue; membrane state lives in d_out (t-1 slice).
//  - Double-buffered LDS with gfx1250 async global->LDS copies (ASYNCcnt).
// ---------------------------------------------------------------------------

typedef __attribute__((ext_vector_type(16))) __bf16       v16bf;
typedef __attribute__((ext_vector_type(8)))  float        v8f;
typedef __attribute__((ext_vector_type(4)))  unsigned int u32x4;
typedef int v4i_vec __attribute__((vector_size(16)));   // matches builtin's pointee

union Frag { v16bf v; u32x4 q[2]; };

#define SNN_BETA 0.95f
#define SNN_THR  1.0f

// ---- gfx1250 async global->LDS copy (falls back to sync copy) --------------
// Probe-derived signature: (v4i addrspace(1)* gsrc, v4i addrspace(3)* ldst,
//                           imm offset, imm cpol)
#if __has_builtin(__builtin_amdgcn_global_load_async_to_lds_b128)
#define HAVE_ASYNC_LDS 1
#define ASYNC_CP_B128(ldsp, glp)                                              \
  __builtin_amdgcn_global_load_async_to_lds_b128(                             \
      (__attribute__((address_space(1))) v4i_vec*)(glp),                      \
      (__attribute__((address_space(3))) v4i_vec*)(ldsp), 0, 0)
#define ASYNC_WAIT() asm volatile("s_wait_asynccnt 0x0" ::: "memory")
#else
#define HAVE_ASYNC_LDS 0
#define ASYNC_CP_B128(ldsp, glp) (*(ldsp) = *(glp))
#define ASYNC_WAIT() ((void)0)
#endif

// ---- fp32 -> (bf16 hi, bf16 lo) split --------------------------------------
__global__ __launch_bounds__(256) void k_split(const float* __restrict__ w,
                                               __bf16* __restrict__ hi,
                                               __bf16* __restrict__ lo,
                                               int n) {
  int i = blockIdx.x * 256 + threadIdx.x;
  if (i < n) {
    float v = w[i];
    __bf16 h = (__bf16)v;            // RNE
    hi[i] = h;
    lo[i] = (__bf16)(v - (float)h);  // residual
  }
}

// ---- per-step layer-0 LIF (GEMM hoisted; cur0 precomputed) -----------------
__global__ __launch_bounds__(256) void k_lif0(const float* __restrict__ cur0,
                                              float* __restrict__ out0,   // [3,T,256,2048]
                                              __bf16* __restrict__ spk_bf,
                                              int T, int t) {
  const size_t NU = (size_t)256 * 2048;
  size_t i = (size_t)blockIdx.x * 256 + threadIdx.x;
  float cur = cur0[i];
  float mp  = (t == 0) ? 0.f : out0[((size_t)T + (t - 1)) * NU + i];
  float mem = SNN_BETA * mp + cur;
  float spk = (mem - SNN_THR) > 0.f ? 1.f : 0.f;
  mem -= spk * SNN_THR;
  out0[(size_t)t * NU + i]            = spk;
  out0[((size_t)T + t) * NU + i]      = mem;
  out0[((size_t)2 * T + t) * NU + i]  = cur;
  spk_bf[i] = (__bf16)spk;
}

// ---- WMMA GEMM (C[256,N] = A[256,K] * W[N,K]^T + b), optional fused LIF ----
// 256 threads = 8 waves; C tile 32(M) x 256(N); wave grid 2x4 -> 16x64/wave.
// K staged 32-deep, double-buffered LDS, 80B row pitch (5 x uint4) so the
// 16-lane b128 fragment reads are bank-conflict-free.
template <int A_HAS_LO, int FUSE>
__global__ __launch_bounds__(256) void k_gemm_lif(
    const __bf16* __restrict__ Ahi, const __bf16* __restrict__ Alo, // [256,K]
    const __bf16* __restrict__ Bhi, const __bf16* __restrict__ Blo, // [N,K]
    const float*  __restrict__ bias,
    float*        __restrict__ cur_out,  // FUSE==0: [256,N]
    float*        __restrict__ outL,     // FUSE==1: [3,T,256,N]
    __bf16*       __restrict__ spk_bf,   // FUSE==1: [256,N]
    int N, int K, int T, int t) {

  __shared__ u32x4 sAh[2 * 32 * 5];
  __shared__ u32x4 sAl[2 * 32 * 5];
  __shared__ u32x4 sBh[2 * 256 * 5];
  __shared__ u32x4 sBl[2 * 256 * 5];

  const int tid   = threadIdx.x;
  const int lane  = tid & 31;
  const int wv    = tid >> 5;
  const int half  = lane >> 4;   // K-half selector (ISA 16-bit A/B layout)
  const int l16   = lane & 15;
  const int m_blk = blockIdx.x * 32;
  const int n_blk = blockIdx.y * 256;
  const int wm    = wv >> 2;     // 0..1
  const int wn    = wv & 3;      // 0..3

  // issue one K-chunk's staging into LDS buffer `buf`
  auto issue_stage = [&](int k0, int buf) {
    if (tid < 128) {
      const int r = tid >> 2, q = tid & 3;
      const u32x4* s = (const u32x4*)(Ahi + (size_t)(m_blk + r) * K + k0);
      ASYNC_CP_B128(&sAh[buf * 160 + r * 5 + q], s + q);
      if constexpr (A_HAS_LO) {
        const u32x4* s2 = (const u32x4*)(Alo + (size_t)(m_blk + r) * K + k0);
        ASYNC_CP_B128(&sAl[buf * 160 + r * 5 + q], s2 + q);
      }
    }
    const u32x4* bh = (const u32x4*)(Bhi + (size_t)(n_blk + tid) * K + k0);
    const u32x4* bl = (const u32x4*)(Blo + (size_t)(n_blk + tid) * K + k0);
#pragma unroll
    for (int q = 0; q < 4; ++q) {
      ASYNC_CP_B128(&sBh[buf * 1280 + tid * 5 + q], bh + q);
      ASYNC_CP_B128(&sBl[buf * 1280 + tid * 5 + q], bl + q);
    }
  };

  v8f acc[4];
#pragma unroll
  for (int i = 0; i < 4; ++i) acc[i] = 0;

  const int nIter = K >> 5;
  issue_stage(0, 0);

  for (int it = 0; it < nIter; ++it) {
    ASYNC_WAIT();        // my async copies into buf[it&1] have landed
    __syncthreads();     // everyone's have; prior reads of buf[(it+1)&1] done
    if (it + 1 < nIter) issue_stage((it + 1) << 5, (it + 1) & 1);

    const int buf = it & 1;
    // A fragment (ISA 7.12.2): lane L -> row L%16; elems 0-7: K=8h..8h+7,
    // elems 8-15: K=16+8h.. -> two b128 LDS reads at byte offs 16h, 32+16h.
    Frag ah, al;
    const int ar = buf * 160 + (wm * 16 + l16) * 5;
    ah.q[0] = sAh[ar + half];
    ah.q[1] = sAh[ar + 2 + half];
    if constexpr (A_HAS_LO) {
      al.q[0] = sAl[ar + half];
      al.q[1] = sAl[ar + 2 + half];
    }

#pragma unroll
    for (int nt = 0; nt < 4; ++nt) {
      const int br = buf * 1280 + (wn * 64 + nt * 16 + l16) * 5;
      Frag bh_, bl_;
      bh_.q[0] = sBh[br + half];
      bh_.q[1] = sBh[br + 2 + half];
      bl_.q[0] = sBl[br + half];
      bl_.q[1] = sBl[br + 2 + half];

      acc[nt] = __builtin_amdgcn_wmma_f32_16x16x32_bf16(
          false, ah.v, false, bh_.v, (short)0, acc[nt], false, false);
      acc[nt] = __builtin_amdgcn_wmma_f32_16x16x32_bf16(
          false, ah.v, false, bl_.v, (short)0, acc[nt], false, false);
      if constexpr (A_HAS_LO)
        acc[nt] = __builtin_amdgcn_wmma_f32_16x16x32_bf16(
            false, al.v, false, bh_.v, (short)0, acc[nt], false, false);
    }
  }

  // epilogue: C layout lane L -> col L%16, VGPR r -> row r + 8*(L/16)
  const size_t slab = (size_t)256 * N;
#pragma unroll
  for (int nt = 0; nt < 4; ++nt) {
    const int col  = n_blk + wn * 64 + nt * 16 + l16;
    const float bc = bias[col];
#pragma unroll
    for (int r = 0; r < 8; ++r) {
      const int row   = m_blk + wm * 16 + r + 8 * half;
      const float cur = acc[nt][r] + bc;
      const size_t pos = (size_t)row * N + col;
      if constexpr (FUSE) {
        float mp  = (t == 0) ? 0.f : outL[((size_t)T + (t - 1)) * slab + pos];
        float mem = SNN_BETA * mp + cur;
        float spk = (mem - SNN_THR) > 0.f ? 1.f : 0.f;
        mem -= spk * SNN_THR;
        outL[(size_t)t * slab + pos]           = spk;
        outL[((size_t)T + t) * slab + pos]     = mem;
        outL[((size_t)2 * T + t) * slab + pos] = cur;
        spk_bf[pos] = (__bf16)spk;
      } else {
        cur_out[pos] = cur;
      }
    }
  }
}

// ---------------------------------------------------------------------------
extern "C" void kernel_launch(void* const* d_in, const int* in_sizes, int n_in,
                              void* d_out, int out_size, void* d_ws, size_t ws_size,
                              hipStream_t stream) {
  const float* x  = (const float*)d_in[0];
  const float* W0 = (const float*)d_in[1];
  const float* b0 = (const float*)d_in[2];
  const float* W1 = (const float*)d_in[3];
  const float* b1 = (const float*)d_in[4];
  const float* W2 = (const float*)d_in[5];
  const float* b2 = (const float*)d_in[6];
  (void)in_sizes; (void)n_in; (void)ws_size;

  const int B = 256, U0 = 1024, U1 = 2048, U2 = 2048, U3 = 1024;
  // num_steps is a device scalar; derive T from out_size (graph-capture safe).
  const int T = out_size / (3 * B * (U1 + U2 + U3));

  char* ws = (char*)d_ws;
  size_t off = 0;
  auto take = [&](size_t bytes) -> char* {
    char* p = ws + off;
    off += (bytes + 255) & ~(size_t)255;
    return p;
  };
  __bf16* w0hi = (__bf16*)take((size_t)U1 * U0 * 2);
  __bf16* w0lo = (__bf16*)take((size_t)U1 * U0 * 2);
  __bf16* w1hi = (__bf16*)take((size_t)U2 * U1 * 2);
  __bf16* w1lo = (__bf16*)take((size_t)U2 * U1 * 2);
  __bf16* w2hi = (__bf16*)take((size_t)U3 * U2 * 2);
  __bf16* w2lo = (__bf16*)take((size_t)U3 * U2 * 2);
  __bf16* xhi  = (__bf16*)take((size_t)B * U0 * 2);
  __bf16* xlo  = (__bf16*)take((size_t)B * U0 * 2);
  float*  cur0 = (float*) take((size_t)B * U1 * 4);
  __bf16* s0bf = (__bf16*)take((size_t)B * U1 * 2);
  __bf16* s1bf = (__bf16*)take((size_t)B * U2 * 2);
  __bf16* s2bf = (__bf16*)take((size_t)B * U3 * 2);  // layer2 spikes (unused downstream)

  // one-time: bf16 hi/lo splits of weights and x
  const int nW0 = U1 * U0, nW1 = U2 * U1, nW2 = U3 * U2, nx = B * U0;
  k_split<<<(nW0 + 255) / 256, 256, 0, stream>>>(W0, w0hi, w0lo, nW0);
  k_split<<<(nW1 + 255) / 256, 256, 0, stream>>>(W1, w1hi, w1lo, nW1);
  k_split<<<(nW2 + 255) / 256, 256, 0, stream>>>(W2, w2hi, w2lo, nW2);
  k_split<<<(nx  + 255) / 256, 256, 0, stream>>>(x,  xhi,  xlo,  nx);

  // one-time: cur0 = x @ W0^T + b0 (loop-invariant across timesteps)
  {
    dim3 g(B / 32, U1 / 256);
    k_gemm_lif<1, 0><<<g, 256, 0, stream>>>(xhi, xlo, w0hi, w0lo, b0,
                                            cur0, nullptr, nullptr,
                                            U1, U0, T, 0);
  }

  float* out0 = (float*)d_out;
  float* out1 = out0 + (size_t)3 * T * B * U1;
  float* out2 = out1 + (size_t)3 * T * B * U2;

  for (int t = 0; t < T; ++t) {
    // layer 0: elementwise LIF on precomputed cur0
    k_lif0<<<(B * U1) / 256, 256, 0, stream>>>(cur0, out0, s0bf, T, t);
    // layer 1: spk0 @ W1^T + b1, fused LIF
    {
      dim3 g(B / 32, U2 / 256);
      k_gemm_lif<0, 1><<<g, 256, 0, stream>>>(s0bf, nullptr, w1hi, w1lo, b1,
                                              nullptr, out1, s1bf,
                                              U2, U1, T, t);
    }
    // layer 2: spk1 @ W2^T + b2, fused LIF
    {
      dim3 g(B / 32, U3 / 256);
      k_gemm_lif<0, 1><<<g, 256, 0, stream>>>(s1bf, nullptr, w2hi, w2lo, b2,
                                              nullptr, out2, s2bf,
                                              U3, U2, T, t);
    }
  }
}
